// EncoderOnly_9148280340576
// MI455X (gfx1250) — matrix operand order repaired and checked by reading, compiled, not run on hardware
//
#include <hip/hip_runtime.h>
#include <stdint.h>

// ---------------------------------------------------------------------------
// 6-layer pre-norm transformer encoder (shared weights) for MI455X / gfx1250.
// - All matmuls via v_wmma_f32_16x16x32_bf16 (fp32 accumulate).
// - GEMM: 128x128 block tile, 8 waves (4x2), 32x64 per wave = 8 WMMA / K-step
//   (fragment reuse across a 2x4 accumulator grid -> matrix-unit bound).
// - A-panel staged to LDS via the Tensor Data Mover (tensor_load_to_lds +
//   s_wait_tensorcnt) overlapped with the transposed B-tile scatter.
// - Fused attention keeps the full 64x1024 fp32 score panel in the 320KB
//   WGP LDS (no score traffic to HBM).
// ---------------------------------------------------------------------------

#define BATCH   8
#define SEQ     1024
#define DMODEL  512
#define NHEAD   8
#define HDIM    64
#define FFDIM   2048
#define NLAYER  6
#define MROWS   (BATCH * SEQ)    /* 8192 */
#define LN_EPS  1e-6f

typedef __attribute__((ext_vector_type(16))) __bf16 v16bf;
typedef __attribute__((ext_vector_type(8)))  float  v8f;

#if defined(__gfx1250__) && __has_builtin(__builtin_amdgcn_tensor_load_to_lds) && \
    __has_builtin(__builtin_amdgcn_s_wait_tensorcnt)
#define USE_TDM 1
typedef __attribute__((ext_vector_type(4))) unsigned int u32x4;
typedef __attribute__((ext_vector_type(8))) int          i32x8;
typedef __attribute__((ext_vector_type(4))) int          i32x4;
#endif

union FragBF { uint32_t u[8]; v16bf v; };

__device__ __forceinline__ uint16_t f2bf(float f) {
  union { float f; uint32_t u; } x; x.f = f;
  uint32_t r = x.u + 0x7FFFu + ((x.u >> 16) & 1u);   // round-to-nearest-even
  return (uint16_t)(r >> 16);
}
__device__ __forceinline__ uint32_t pack2bf(float a, float b) {
  return (uint32_t)f2bf(a) | ((uint32_t)f2bf(b) << 16);
}

// ---------------------------------------------------------------------------
// fp32 -> bf16 weight conversion
// ---------------------------------------------------------------------------
__global__ void cvt_f32_bf16(const float* __restrict__ src,
                             uint16_t* __restrict__ dst, int n) {
  int i = blockIdx.x * blockDim.x + threadIdx.x;
  if (i < n) dst[i] = f2bf(src[i]);
}

// ---------------------------------------------------------------------------
// Embedding gather * sqrt(D) + sinusoidal positional encoding -> x (fp32)
// ---------------------------------------------------------------------------
__global__ void embed_pe(const int* __restrict__ tok,
                         const float* __restrict__ emb,
                         float* __restrict__ x) {
  const int row = blockIdx.x;            // 0..MROWS-1  (= b*SEQ + s)
  const int s   = row % SEQ;
  const int t   = tok[row];
  const float sq = 22.62741699796952f;   // sqrt(512)
  for (int d = threadIdx.x; d < DMODEL; d += blockDim.x) {
    int i = d >> 1;
    float div = __expf((float)(2 * i) * (-9.210340371976184f / (float)DMODEL));
    float ang = (float)s * div;
    float pe  = (d & 1) ? __cosf(ang) : __sinf(ang);
    x[(size_t)row * DMODEL + d] = emb[(size_t)t * DMODEL + d] * sq + pe;
  }
}

// ---------------------------------------------------------------------------
// Custom LayerNorm (mean, *unbiased* std, eps added to std, scalar a/b),
// fp32 in -> bf16 out.  One 256-thread block per row (D = 512).
// ---------------------------------------------------------------------------
__global__ __launch_bounds__(256)
void layernorm_bf16(const float* __restrict__ x,
                    const float* __restrict__ alpha,
                    const float* __restrict__ beta,
                    uint16_t* __restrict__ out) {
  __shared__ float sred[2][8];
  const int row = blockIdx.x;
  const float* xr = x + (size_t)row * DMODEL;
  float v0 = xr[threadIdx.x];
  float v1 = xr[threadIdx.x + 256];
  float s  = v0 + v1;
  float sq = v0 * v0 + v1 * v1;
  for (int off = 16; off > 0; off >>= 1) {        // wave32 reduction
    s  += __shfl_down(s,  off, 32);
    sq += __shfl_down(sq, off, 32);
  }
  const int lane = threadIdx.x & 31, wave = threadIdx.x >> 5;
  if (lane == 0) { sred[0][wave] = s; sred[1][wave] = sq; }
  __syncthreads();
  float ts = 0.f, tq = 0.f;
  #pragma unroll
  for (int w = 0; w < 8; ++w) { ts += sred[0][w]; tq += sred[1][w]; }
  const float mean = ts / (float)DMODEL;
  const float var  = (tq - (float)DMODEL * mean * mean) / (float)(DMODEL - 1);
  const float inv  = alpha[0] / (sqrtf(fmaxf(var, 0.f)) + LN_EPS);
  const float b    = beta[0];
  out[(size_t)row * DMODEL + threadIdx.x]       = f2bf((v0 - mean) * inv + b);
  out[(size_t)row * DMODEL + threadIdx.x + 256] = f2bf((v1 - mean) * inv + b);
}

// ---------------------------------------------------------------------------
// WMMA GEMM: C[M,N] = epilogue(A[M,K](bf16) * W[K,N](bf16) + bias)
//   RELU   : relu before store
//   RESID  : Cf (fp32) += result  (residual accumulate into x)
//   OUTBF  : Cb (bf16)  = result
// 128x128 block tile, 8 waves (4 M-groups x 2 N-groups), 2x4 accums/wave,
// K-step 32.  A tile via TDM DMA (when available), B tile transposed scatter.
// ---------------------------------------------------------------------------
template <bool RELU, bool RESID, bool OUTBF>
__global__ __launch_bounds__(256)
void gemm_wmma(const uint16_t* __restrict__ A, const uint16_t* __restrict__ W,
               const float* __restrict__ bias, int K, int N,
               float* __restrict__ Cf, uint16_t* __restrict__ Cb) {
  __shared__ uint16_t ldsA[128 * 32];   // [m][k]   8KB
  __shared__ uint16_t ldsBt[128 * 32];  // [n][k]   8KB (W tile transposed)
  const int tid  = threadIdx.x;
  const int lane = tid & 31, wave = tid >> 5;
  const int wm   = wave & 3, wn = wave >> 2;      // 4 x 2 wave grid
  const int l16  = lane & 15, hs = lane >> 4;
  const int mBase = blockIdx.x * 128;
  const int nBase = blockIdx.y * 128;
  v8f acc[2][4] = {};

  for (int k0 = 0; k0 < K; k0 += 32) {
#ifdef USE_TDM
    if (wave == 0) {
      // Tensor Data Mover: 2D tile, 128 rows x 32 bf16, row stride = K elems.
      const uint64_t ga = (uint64_t)(uintptr_t)(A + (size_t)mBase * K + k0);
      const uint32_t la = (uint32_t)(uintptr_t)(void*)ldsA;   // LDS byte offset
      u32x4 g0;
      g0[0] = 1u;                                          // count=1 (valid D#)
      g0[1] = la;                                          // lds_addr
      g0[2] = (uint32_t)ga;                                // global_addr[31:0]
      g0[3] = (uint32_t)((ga >> 32) & 0x01FFFFFFu) | (2u << 30); // addr[56:32] | type=2
      i32x8 g1;
      g1[0] = (int)(1u << 16);                             // data_size=1 (2B), no mcast
      g1[1] = (int)((uint32_t)K << 16);                    // tensor_dim0 lo16
      g1[2] = (int)(((uint32_t)K >> 16) | ((uint32_t)MROWS << 16)); // dim0 hi | dim1 lo
      g1[3] = (int)(((uint32_t)MROWS >> 16) | (32u << 16));         // dim1 hi | tile_dim0=32
      g1[4] = 128;                                         // tile_dim1=128, tile_dim2=0
      g1[5] = K;                                           // tensor_dim0_stride lo32
      g1[6] = 0;                                           // stride hi | dim1_stride lo
      g1[7] = 0;
      i32x4 gz4 = {0, 0, 0, 0};                            // groups 2/3 unused (2D)
      i32x8 gz8 = {0, 0, 0, 0, 0, 0, 0, 0};
      __builtin_amdgcn_tensor_load_to_lds(g0, g1, gz4, gz4, gz8, 0);
    }
#else
    { // A tile 128x32 manual: 2x 16B loads per thread
      int r = tid >> 1, g = tid & 1;
      const uint16_t* src = A + (size_t)(mBase + r) * K + k0 + g * 16;
      *(uint4*)(ldsA + r * 32 + g * 16)     = *(const uint4*)(src);
      *(uint4*)(ldsA + r * 32 + g * 16 + 8) = *(const uint4*)(src + 8);
    }
#endif
    { // W tile 32x128 -> LDS transposed [n][k]  (overlaps with the TDM DMA)
      int kr = tid >> 3, g = tid & 7;
      const uint16_t* src = W + (size_t)(k0 + kr) * N + nBase + g * 16;
      uint4 d0 = *(const uint4*)(src);
      uint4 d1 = *(const uint4*)(src + 8);
      const uint16_t* e0 = (const uint16_t*)&d0;
      const uint16_t* e1 = (const uint16_t*)&d1;
      #pragma unroll
      for (int j = 0; j < 8; ++j) {
        ldsBt[(g * 16 + j) * 32 + kr]     = e0[j];
        ldsBt[(g * 16 + 8 + j) * 32 + kr] = e1[j];
      }
    }
    if (k0 + 32 < K)   // prefetch next W K-slice
      __builtin_prefetch(W + (size_t)(k0 + 32 + (tid >> 3)) * N + nBase, 0, 1);
#ifdef USE_TDM
    if (wave == 0) __builtin_amdgcn_s_wait_tensorcnt(0);
#endif
    __syncthreads();

    FragBF a[2], bfr[4];
    #pragma unroll
    for (int t = 0; t < 2; ++t)
      #pragma unroll
      for (int vr = 0; vr < 8; ++vr) {   // A frag: lane=M, K pairs per ISA layout
        int gg = vr >> 2, rr = vr & 3;
        int kk = gg * 16 + hs * 8 + rr * 2;
        a[t].u[vr] = *(const uint32_t*)(ldsA + (wm * 32 + t * 16 + l16) * 32 + kk);
      }
    #pragma unroll
    for (int j = 0; j < 4; ++j)
      #pragma unroll
      for (int vr = 0; vr < 8; ++vr) {   // B frag: lane=N, K split across halfwave
        int kk = (vr >> 2) * 16 + hs * 8 + (vr & 3) * 2;
        bfr[j].u[vr] = *(const uint32_t*)(ldsBt + (wn * 64 + j * 16 + l16) * 32 + kk);
      }
    #pragma unroll
    for (int t = 0; t < 2; ++t)
      #pragma unroll
      for (int j = 0; j < 4; ++j)
        acc[t][j] = __builtin_amdgcn_wmma_f32_16x16x32_bf16(
            false, a[t].v, false, bfr[j].v, (short)0, acc[t][j], false, false);
    __syncthreads();
  }

  #pragma unroll
  for (int t = 0; t < 2; ++t)
    #pragma unroll
    for (int j = 0; j < 4; ++j)
      #pragma unroll
      for (int i = 0; i < 8; ++i) {      // C layout: VGPR i -> M = i + 8*(lane/16)
        int m = mBase + wm * 32 + t * 16 + i + hs * 8;
        int n = nBase + wn * 64 + j * 16 + l16;
        float xv = acc[t][j][i] + bias[n];
        if (RELU) xv = fmaxf(xv, 0.f);
        if (RESID)      Cf[(size_t)m * N + n] += xv;
        else if (OUTBF) Cb[(size_t)m * N + n]  = f2bf(xv);
        else            Cf[(size_t)m * N + n]  = xv;
      }
}

// ---------------------------------------------------------------------------
// Fused attention for one (b, h, 64-query tile).
// Full 64x1024 fp32 score panel kept in LDS (256KB of the WGP's 320KB):
//   phase 1: scores = scale * Q K^T (+mask) via WMMA, K tiles streamed via LDS
//   phase 2: exact softmax in LDS
//   phase 3: O = P V via WMMA, V tiles staged transposed in LDS
// ---------------------------------------------------------------------------
__global__ __launch_bounds__(256, 1)
void attention_wmma(const uint16_t* __restrict__ Q, const uint16_t* __restrict__ Kb,
                    const uint16_t* __restrict__ V, const int* __restrict__ mask,
                    uint16_t* __restrict__ O) {
  extern __shared__ char smem[];
  float*    sc  = (float*)smem;                                   // [64][1024]
  uint16_t* qt  = (uint16_t*)(smem + 64 * SEQ * 4);               // [64][64]
  uint16_t* kt  = (uint16_t*)(smem + 64 * SEQ * 4 + 64 * 64 * 2); // [64][64]
  float*    red = (float*)(smem + 64 * SEQ * 4 + 2 * 64 * 64 * 2);// [64][4]

  const int tid  = threadIdx.x;
  const int lane = tid & 31, wave = tid >> 5;
  const int wm   = wave & 3, wn = wave >> 2;
  const int l16  = lane & 15, hs = lane >> 4;
  const int bh   = blockIdx.x;
  const int b    = bh / NHEAD, h = bh % NHEAD;
  const int qBase = blockIdx.y * 64;
  const size_t base = ((size_t)b * SEQ) * DMODEL + (size_t)h * HDIM;

  // ---- load Q tile [64 q][64 d] into LDS, build per-wave A fragments ----
  {
    int r = tid >> 3, g = tid & 7;
    #pragma unroll
    for (int rep = 0; rep < 2; ++rep) {
      int rr = r + rep * 32;
      const uint4 d4 = *(const uint4*)(Q + base + (size_t)(qBase + rr) * DMODEL + g * 8);
      *(uint4*)(qt + rr * 64 + g * 8) = d4;
    }
  }
  __syncthreads();
  FragBF qf[2];
  #pragma unroll
  for (int ks = 0; ks < 2; ++ks)
    #pragma unroll
    for (int vr = 0; vr < 8; ++vr) {
      int gg = vr >> 2, rr = vr & 3;
      int kk = ks * 32 + gg * 16 + hs * 8 + rr * 2;
      qf[ks].u[vr] = *(const uint32_t*)(qt + (wm * 16 + l16) * 64 + kk);
    }

  const float scale = 0.125f;   // 1/sqrt(HD=64)

  // ---- phase 1: score panel ----
  for (int j = 0; j < SEQ / 64; ++j) {
    { // K tile [64 kpos][64 d]
      int r = tid >> 3, g = tid & 7;
      #pragma unroll
      for (int rep = 0; rep < 2; ++rep) {
        int rr = r + rep * 32;
        const uint4 d4 = *(const uint4*)(Kb + base + (size_t)(j * 64 + rr) * DMODEL + g * 8);
        *(uint4*)(kt + rr * 64 + g * 8) = d4;
      }
    }
    __syncthreads();
    v8f s0 = {}, s1 = {};
    #pragma unroll
    for (int ks = 0; ks < 2; ++ks) {
      FragBF b0, b1;   // B = K^T: contraction=d (contiguous in kt rows), col=kpos
      #pragma unroll
      for (int vr = 0; vr < 8; ++vr) {
        int kk = ks * 32 + (vr >> 2) * 16 + hs * 8 + (vr & 3) * 2;
        b0.u[vr] = *(const uint32_t*)(kt + (wn * 32 + l16) * 64 + kk);
        b1.u[vr] = *(const uint32_t*)(kt + (wn * 32 + 16 + l16) * 64 + kk);
      }
      s0 = __builtin_amdgcn_wmma_f32_16x16x32_bf16(false, qf[ks].v, false, b0.v,
                                                   (short)0, s0, false, false);
      s1 = __builtin_amdgcn_wmma_f32_16x16x32_bf16(false, qf[ks].v, false, b1.v,
                                                   (short)0, s1, false, false);
    }
    #pragma unroll
    for (int i = 0; i < 8; ++i) {
      int m  = wm * 16 + i + hs * 8;
      int c0 = j * 64 + wn * 32 + l16;
      int c1 = c0 + 16;
      float v0s = s0[i] * scale, v1s = s1[i] * scale;
      if (mask[b * SEQ + c0] == 0) v0s = -1e9f;
      if (mask[b * SEQ + c1] == 0) v1s = -1e9f;
      sc[m * SEQ + c0] = v0s;
      sc[m * SEQ + c1] = v1s;
    }
    __syncthreads();
  }

  // ---- phase 2: exact softmax, 4 threads per row ----
  {
    int row = tid >> 2, sub = tid & 3;
    float* srow = sc + row * SEQ;
    float mx = -3.4e38f;
    for (int c = sub; c < SEQ; c += 4) mx = fmaxf(mx, srow[c]);
    red[row * 4 + sub] = mx;
    __syncthreads();
    float m4 = fmaxf(fmaxf(red[row * 4 + 0], red[row * 4 + 1]),
                     fmaxf(red[row * 4 + 2], red[row * 4 + 3]));
    float sum = 0.f;
    for (int c = sub; c < SEQ; c += 4) {
      float e = __expf(srow[c] - m4);
      srow[c] = e;
      sum += e;
    }
    __syncthreads();
    red[row * 4 + sub] = sum;
    __syncthreads();
    float inv = 1.f / (red[row * 4 + 0] + red[row * 4 + 1] +
                       red[row * 4 + 2] + red[row * 4 + 3]);
    for (int c = sub; c < SEQ; c += 4) srow[c] *= inv;
  }
  __syncthreads();

  // ---- phase 3: O = P * V ----
  v8f o0 = {}, o1 = {};
  for (int j = 0; j < SEQ / 64; ++j) {
    { // V tile staged transposed: kt[d][kpos]
      int r = tid >> 3, g = tid & 7;
      #pragma unroll
      for (int rep = 0; rep < 2; ++rep) {
        int kp = r + rep * 32;
        const uint4 d4 = *(const uint4*)(V + base + (size_t)(j * 64 + kp) * DMODEL + g * 8);
        const uint16_t* e = (const uint16_t*)&d4;
        #pragma unroll
        for (int q2 = 0; q2 < 8; ++q2)
          kt[(g * 8 + q2) * 64 + kp] = e[q2];
      }
    }
    __syncthreads();
    #pragma unroll
    for (int ks = 0; ks < 2; ++ks) {
      FragBF pa;   // A = probs, fp32 in LDS -> packed bf16 pairs
      #pragma unroll
      for (int vr = 0; vr < 8; ++vr) {
        int gg = vr >> 2, rr = vr & 3;
        int kk = j * 64 + ks * 32 + gg * 16 + hs * 8 + rr * 2;
        const float* p = sc + (wm * 16 + l16) * SEQ + kk;
        pa.u[vr] = pack2bf(p[0], p[1]);
      }
      FragBF b0, b1;  // B = V: contraction=kpos (contiguous in kt rows), col=d
      #pragma unroll
      for (int vr = 0; vr < 8; ++vr) {
        int kk = ks * 32 + (vr >> 2) * 16 + hs * 8 + (vr & 3) * 2;
        b0.u[vr] = *(const uint32_t*)(kt + (wn * 32 + l16) * 64 + kk);
        b1.u[vr] = *(const uint32_t*)(kt + (wn * 32 + 16 + l16) * 64 + kk);
      }
      o0 = __builtin_amdgcn_wmma_f32_16x16x32_bf16(false, pa.v, false, b0.v,
                                                   (short)0, o0, false, false);
      o1 = __builtin_amdgcn_wmma_f32_16x16x32_bf16(false, pa.v, false, b1.v,
                                                   (short)0, o1, false, false);
    }
    __syncthreads();
  }

  #pragma unroll
  for (int i = 0; i < 8; ++i) {
    int m  = wm * 16 + i + hs * 8;
    int n0 = wn * 32 + l16, n1 = n0 + 16;
    size_t orow = base + (size_t)(qBase + m) * DMODEL;
    O[orow + n0] = f2bf(o0[i]);
    O[orow + n1] = f2bf(o1[i]);
  }
}

// ---------------------------------------------------------------------------
// Host-side orchestration
// ---------------------------------------------------------------------------
extern "C" void kernel_launch(void* const* d_in, const int* in_sizes, int n_in,
                              void* d_out, int out_size, void* d_ws, size_t ws_size,
                              hipStream_t stream) {
  const int*   tokens = (const int*)d_in[0];
  const int*   maskp  = (const int*)d_in[1];
  const float* emb    = (const float*)d_in[2];
  const float* Wq = (const float*)d_in[3];  const float* bq = (const float*)d_in[4];
  const float* Wk = (const float*)d_in[5];  const float* bk = (const float*)d_in[6];
  const float* Wv = (const float*)d_in[7];  const float* bv = (const float*)d_in[8];
  const float* Wo = (const float*)d_in[9];  const float* bo = (const float*)d_in[10];
  const float* W1 = (const float*)d_in[11]; const float* b1 = (const float*)d_in[12];
  const float* W2 = (const float*)d_in[13]; const float* b2 = (const float*)d_in[14];
  const float* ln_a1 = (const float*)d_in[15]; const float* ln_b1 = (const float*)d_in[16];
  const float* ln_a2 = (const float*)d_in[17]; const float* ln_b2 = (const float*)d_in[18];

  float* x = (float*)d_out;                       // [B,S,D] fp32, lives in d_out

  char* ws = (char*)d_ws;
  uint16_t* hN  = (uint16_t*)ws; ws += (size_t)MROWS * DMODEL * 2;
  uint16_t* qB  = (uint16_t*)ws; ws += (size_t)MROWS * DMODEL * 2;
  uint16_t* kB  = (uint16_t*)ws; ws += (size_t)MROWS * DMODEL * 2;
  uint16_t* vB  = (uint16_t*)ws; ws += (size_t)MROWS * DMODEL * 2;
  uint16_t* oB  = (uint16_t*)ws; ws += (size_t)MROWS * DMODEL * 2;
  uint16_t* f1  = (uint16_t*)ws; ws += (size_t)MROWS * FFDIM * 2;
  uint16_t* WqB = (uint16_t*)ws; ws += (size_t)DMODEL * DMODEL * 2;
  uint16_t* WkB = (uint16_t*)ws; ws += (size_t)DMODEL * DMODEL * 2;
  uint16_t* WvB = (uint16_t*)ws; ws += (size_t)DMODEL * DMODEL * 2;
  uint16_t* WoB = (uint16_t*)ws; ws += (size_t)DMODEL * DMODEL * 2;
  uint16_t* W1B = (uint16_t*)ws; ws += (size_t)DMODEL * FFDIM * 2;
  uint16_t* W2B = (uint16_t*)ws; ws += (size_t)FFDIM * DMODEL * 2;

  const int nDD = DMODEL * DMODEL, nDF = DMODEL * FFDIM;
  cvt_f32_bf16<<<(nDD + 255) / 256, 256, 0, stream>>>(Wq, WqB, nDD);
  cvt_f32_bf16<<<(nDD + 255) / 256, 256, 0, stream>>>(Wk, WkB, nDD);
  cvt_f32_bf16<<<(nDD + 255) / 256, 256, 0, stream>>>(Wv, WvB, nDD);
  cvt_f32_bf16<<<(nDD + 255) / 256, 256, 0, stream>>>(Wo, WoB, nDD);
  cvt_f32_bf16<<<(nDF + 255) / 256, 256, 0, stream>>>(W1, W1B, nDF);
  cvt_f32_bf16<<<(nDF + 255) / 256, 256, 0, stream>>>(W2, W2B, nDF);

  embed_pe<<<MROWS, 256, 0, stream>>>(tokens, emb, x);

  const dim3 gP (MROWS / 128, DMODEL / 128);   // 64 x 4
  const dim3 gF1(MROWS / 128, FFDIM / 128);    // 64 x 16
  const dim3 gAtt(BATCH * NHEAD, SEQ / 64);    // 64 x 16
  const size_t attLds = (size_t)64 * SEQ * 4 + 2 * 64 * 64 * 2 + 64 * 4 * 4;

  for (int l = 0; l < NLAYER; ++l) {
    layernorm_bf16<<<MROWS, 256, 0, stream>>>(x, ln_a1, ln_b1, hN);
    gemm_wmma<false, false, true><<<gP, 256, 0, stream>>>(hN, WqB, bq, DMODEL, DMODEL, nullptr, qB);
    gemm_wmma<false, false, true><<<gP, 256, 0, stream>>>(hN, WkB, bk, DMODEL, DMODEL, nullptr, kB);
    gemm_wmma<false, false, true><<<gP, 256, 0, stream>>>(hN, WvB, bv, DMODEL, DMODEL, nullptr, vB);
    attention_wmma<<<gAtt, 256, attLds, stream>>>(qB, kB, vB, maskp, oB);
    gemm_wmma<false, true, false><<<gP, 256, 0, stream>>>(oB, WoB, bo, DMODEL, DMODEL, x, nullptr);
    layernorm_bf16<<<MROWS, 256, 0, stream>>>(x, ln_a2, ln_b2, hN);
    gemm_wmma<true, false, true><<<gF1, 256, 0, stream>>>(hN, W1B, b1, DMODEL, FFDIM, nullptr, f1);
    gemm_wmma<false, true, false><<<gP, 256, 0, stream>>>(f1, W2B, b2, FFDIM, DMODEL, x, nullptr);
  }
}
